// OrthogonalBivectorBlock_28741921144993
// MI455X (gfx1250) — compile-verified, matrix operand order checked by reading, not run on hardware
//
#include <hip/hip_runtime.h>
#include <hip/hip_bf16.h>
#include <math.h>

// ---------------------------------------------------------------------------
// MI455X (gfx1250, wave32) implementation.
// Pipeline:
//  P0  convert x -> f16, transpose-convert all weights -> f16 [N,K]
//  K1  Hk = gelu(x@kw1+b), Hq = gelu(x@qw1+b), Vt = (x@vw+b)^T     (WMMA f16,
//      B tile staged in LDS via async-load-to-LDS, ping-pong)
//  K2  phases -> Q[l,f]=w_s*{cos,sin}(qp), K[l,f], Kt[f,l]          (WMMA f16)
//  K3a per-chunk state sums S_j = K_j^T V_j                         (WMMA f16)
//  K3b exclusive prefix over 16 chunks -> Mt (transposed, f16)      (scan)
//  K3c O = Q@M_prev + tril(Q K^T) @ V  (scores staged in LDS)       (WMMA f16)
//  K4a position-norm + LayerNorm -> yn (f16)
//  K4b out = x + yn@ow + ob  (async LDS-staged B, like K1)          (WMMA f16)
// All accumulation in f32 (WMMA f32 accumulators).
// ---------------------------------------------------------------------------

typedef __attribute__((ext_vector_type(16))) _Float16 v16h;
typedef __attribute__((ext_vector_type(8)))  _Float16 v8h;
typedef __attribute__((ext_vector_type(8)))  float    v8f;
typedef __attribute__((ext_vector_type(4)))  int      v4i;

#define BATCH 2
#define SEQ   2048
#define DIM   512
#define NSET  4
#define NPLN  16
#define TP    64          // NSET*NPLN
#define FDIM  128         // 2*TP (cos|sin features)
#define ROWS  4096        // BATCH*SEQ
#define CHUNK 128
#define NCHK  16          // SEQ/CHUNK
#define PI_F  3.14159265358979323846f

// CDNA5 async global->LDS path (ASYNCcnt). Guarded so the file compiles on
// toolchains without the builtins (falls back to sync reg->LDS staging).
#if defined(__has_builtin)
#if __has_builtin(__builtin_amdgcn_global_load_async_to_lds_b128) && \
    __has_builtin(__builtin_amdgcn_s_wait_asynccnt)
#define USE_ASYNC_LDS 1
#endif
#endif

#define AS_GLOBAL __attribute__((address_space(1)))
#define AS_LDS    __attribute__((address_space(3)))

// ---- WMMA helpers ---------------------------------------------------------

__device__ __forceinline__ v8f wmma_f16(v16h a, v16h b, v8f c) {
  // v_wmma_f32_16x16x32_f16 : D = A(16x32) x B(32x16) + C
  return __builtin_amdgcn_wmma_f32_16x16x32_f16(false, a, false, b,
                                                (short)0, c, false, false);
}

// Load a 16x32 f16 A-fragment (or the transposed-B fragment, which has the
// identical per-lane layout) from a row-major f16 matrix with leading dim ld.
// ISA layout: lane&15 selects row, (lane>>4)*8 selects K phase; vector
// elements 0..7 hold K = kb..kb+7, elements 8..15 hold K = kb+16..kb+23.
__device__ __forceinline__ v16h load_frag(const _Float16* __restrict__ base,
                                          int ld, int rowBase, int k0, int lane) {
  const _Float16* p = base + (size_t)(rowBase + (lane & 15)) * ld
                           + k0 + ((lane >> 4) << 3);
  v16h f;
  *(v8h*)&f       = *(const v8h*)(p);
  *((v8h*)&f + 1) = *(const v8h*)(p + 16);
  return f;
}

// Stage one 64x32 f16 B sub-tile (4 KB) into LDS, one b128 per thread of a
// 256-thread block. Async (ASYNCcnt path) when available.
__device__ __forceinline__ void stage_b_tile(_Float16* __restrict__ lds_dst,
                                             const _Float16* __restrict__ gsrc,
                                             int ld, int colBase, int k0, int tid) {
  const int r = tid >> 2;          // 0..63  (row of transposed B)
  const int c = (tid & 3) << 3;    // 0,8,16,24 (halves)
  const _Float16* src = gsrc + (size_t)(colBase + r) * ld + k0 + c;
  _Float16* dst = lds_dst + r * 32 + c;
#if defined(USE_ASYNC_LDS)
  _Float16* src_nc = const_cast<_Float16*>(src);
  __builtin_amdgcn_global_load_async_to_lds_b128(
      (AS_GLOBAL v4i*)src_nc, (AS_LDS v4i*)dst, 0, 0);
#else
  *(v8h*)dst = *(const v8h*)src;
#endif
}

__device__ __forceinline__ void stage_wait() {
#if defined(USE_ASYNC_LDS)
  __builtin_amdgcn_s_wait_asynccnt(0);
#endif
}

__device__ __forceinline__ float gelu_exact(float v) {
  return 0.5f * v * (1.0f + erff(v * 0.70710678118654752440f));
}

// ---- P0: conversions ------------------------------------------------------

__global__ void k_cvt_f32_to_f16(const float* __restrict__ src,
                                 _Float16* __restrict__ dst, int n) {
  int i = blockIdx.x * blockDim.x + threadIdx.x;
  if (i < n) dst[i] = (_Float16)src[i];
}

// dst[n*K+k] = (f16) src[k*N+n]   (store weights transposed for B operands)
__global__ void k_transpose_f16(const float* __restrict__ src,
                                _Float16* __restrict__ dst, int K, int N) {
  int i = blockIdx.x * blockDim.x + threadIdx.x;
  if (i >= K * N) return;
  int k = i / N, n = i - k * N;
  dst[(size_t)n * K + k] = (_Float16)src[i];
}

// ---- K1: encoder layer-1 GEMMs + V projection -----------------------------
// grid (32, 8, 3), block 256 (8 waves); 128x64 output tile per block.
// B tile ping-pong staged in LDS (async), A fragments double-buffered in regs.

__global__ void k_encode1(const _Float16* __restrict__ xh,
                          const _Float16* __restrict__ kw1t,
                          const _Float16* __restrict__ qw1t,
                          const _Float16* __restrict__ vwt,
                          const float* __restrict__ kb1,
                          const float* __restrict__ qb1,
                          const float* __restrict__ vb,
                          _Float16* __restrict__ hk,
                          _Float16* __restrict__ hq,
                          _Float16* __restrict__ vt) {
  __shared__ _Float16 bs[2][64 * 32];  // 8 KB ping-pong B tile
  const int tid = threadIdx.x;
  const int lane = tid & 31, wave = tid >> 5;
  const int z = blockIdx.z;
  const int rowBase = blockIdx.x * 128 + wave * 16;
  const int colBase = blockIdx.y * 64;
  const _Float16* wt  = (z == 0) ? kw1t : (z == 1) ? qw1t : vwt;
  const float*    bia = (z == 0) ? kb1  : (z == 1) ? qb1  : vb;

  stage_b_tile(bs[0], wt, DIM, colBase, 0, tid);
  v16h a_cur = load_frag(xh, DIM, rowBase, 0, lane);

  v8f acc[4] = {};
  for (int i = 0; i < DIM / 32; ++i) {
    stage_wait();
    __syncthreads();                      // buffer i&1 ready for all waves
    if (i + 1 < DIM / 32)
      stage_b_tile(bs[(i + 1) & 1], wt, DIM, colBase, (i + 1) * 32, tid);
    v16h a_next;
    if (i + 1 < DIM / 32)
      a_next = load_frag(xh, DIM, rowBase, (i + 1) * 32, lane);
#pragma unroll
    for (int t = 0; t < 4; ++t) {
      v16h b = load_frag(bs[i & 1], 32, t * 16, 0, lane);
      acc[t] = wmma_f16(a_cur, b, acc[t]);
    }
    a_cur = a_next;
  }

  const int mBase = rowBase + ((lane >> 4) << 3);
#pragma unroll
  for (int t = 0; t < 4; ++t) {
    const int n = colBase + t * 16 + (lane & 15);
#pragma unroll
    for (int r = 0; r < 8; ++r) {
      const int m = mBase + r;
      float v = acc[t][r] + bia[n];
      if (z < 2) {
        v = gelu_exact(v);
        _Float16* dst = (z == 0) ? hk : hq;
        dst[(size_t)m * DIM + n] = (_Float16)v;
      } else {
        const int b = m >> 11, l = m & 2047;
        vt[((size_t)(b * DIM + n)) * SEQ + l] = (_Float16)v;  // V^T [b][d][l]
      }
    }
  }
}

// ---- K2: encoder layer-2 + phases -> Q, K, Kt -----------------------------
// grid (32), block 256. N = TP = 64 (4 tiles) for each of k/q.

__global__ void k_phases(const _Float16* __restrict__ hk,
                         const _Float16* __restrict__ hq,
                         const _Float16* __restrict__ kw2t,
                         const _Float16* __restrict__ qw2t,
                         const float* __restrict__ kb2,
                         const float* __restrict__ qb2,
                         const float* __restrict__ sw,
                         _Float16* __restrict__ Qm,   // [ROWS][FDIM]
                         _Float16* __restrict__ Km,   // [ROWS][FDIM]
                         _Float16* __restrict__ Kt) { // [B][FDIM][SEQ]
  const int lane = threadIdx.x & 31, wave = threadIdx.x >> 5;
  const int rowBase = blockIdx.x * 128 + wave * 16;

  v8f ak[4] = {}, aq[4] = {};
  for (int k0 = 0; k0 < DIM; k0 += 32) {
    v16h a0 = load_frag(hk, DIM, rowBase, k0, lane);
    v16h a1 = load_frag(hq, DIM, rowBase, k0, lane);
#pragma unroll
    for (int t = 0; t < 4; ++t) {
      v16h bk = load_frag(kw2t, DIM, t * 16, k0, lane);
      v16h bq = load_frag(qw2t, DIM, t * 16, k0, lane);
      ak[t] = wmma_f16(a0, bk, ak[t]);
      aq[t] = wmma_f16(a1, bq, aq[t]);
    }
  }
  // softmax over 4 set weights (redundant per thread; trivially cheap)
  float s0 = sw[0], s1 = sw[1], s2 = sw[2], s3 = sw[3];
  float mx = fmaxf(fmaxf(s0, s1), fmaxf(s2, s3));
  float e0 = expf(s0 - mx), e1 = expf(s1 - mx), e2 = expf(s2 - mx), e3 = expf(s3 - mx);
  float inv = 1.0f / (e0 + e1 + e2 + e3);
  float wsm[4] = {e0 * inv, e1 * inv, e2 * inv, e3 * inv};

  const int mBase = rowBase + ((lane >> 4) << 3);
#pragma unroll
  for (int t = 0; t < 4; ++t) {
    const int tp = t * 16 + (lane & 15);   // 0..63
    const int s  = tp >> 4;
#pragma unroll
    for (int r = 0; r < 8; ++r) {
      const int row = mBase + r;
      const int b = row >> 11, l = row & 2047;
      float kph = tanhf(ak[t][r] + kb2[tp]) * PI_F;
      float qph = tanhf(aq[t][r] + qb2[tp]) * PI_F;
      float kc = cosf(kph), ks = sinf(kph);
      float qc = wsm[s] * cosf(qph), qs = wsm[s] * sinf(qph);
      Km[(size_t)row * FDIM + tp]      = (_Float16)kc;
      Km[(size_t)row * FDIM + TP + tp] = (_Float16)ks;
      Kt[((size_t)(b * FDIM + tp)) * SEQ + l]      = (_Float16)kc;
      Kt[((size_t)(b * FDIM + TP + tp)) * SEQ + l] = (_Float16)ks;
      Qm[(size_t)row * FDIM + tp]      = (_Float16)qc;
      Qm[(size_t)row * FDIM + TP + tp] = (_Float16)qs;
    }
  }
}

// ---- K3a: per-chunk KV sums S_j[f,d] = sum_t K[t,f] V[t,d] ---------------
// grid (NCHK, 8, BATCH), block 256. Register double-buffered K loop.

__global__ void k_chunk_kv(const _Float16* __restrict__ Kt,
                           const _Float16* __restrict__ Vt,
                           float* __restrict__ Ssum) { // [B][NCHK][FDIM][DIM]
  const int lane = threadIdx.x & 31, wave = threadIdx.x >> 5;
  const int j = blockIdx.x, dBase = blockIdx.y * 64, b = blockIdx.z;
  const int fBase = wave * 16;
  const _Float16* ktb = Kt + (size_t)b * FDIM * SEQ;
  const _Float16* vtb = Vt + (size_t)b * DIM * SEQ;
  const int kBase = j * CHUNK;

  v8f acc[4] = {};
  v16h a_cur = load_frag(ktb, SEQ, fBase, kBase, lane);
  v16h b_cur[4];
#pragma unroll
  for (int t = 0; t < 4; ++t)
    b_cur[t] = load_frag(vtb, SEQ, dBase + t * 16, kBase, lane);

  for (int i = 0; i < CHUNK / 32; ++i) {
    v16h a_next, b_next[4];
    if (i + 1 < CHUNK / 32) {
      a_next = load_frag(ktb, SEQ, fBase, kBase + (i + 1) * 32, lane);
#pragma unroll
      for (int t = 0; t < 4; ++t)
        b_next[t] = load_frag(vtb, SEQ, dBase + t * 16, kBase + (i + 1) * 32, lane);
    }
#pragma unroll
    for (int t = 0; t < 4; ++t) acc[t] = wmma_f16(a_cur, b_cur[t], acc[t]);
    if (i + 1 < CHUNK / 32) {
      a_cur = a_next;
#pragma unroll
      for (int t = 0; t < 4; ++t) b_cur[t] = b_next[t];
    }
  }

  const int fB = fBase + ((lane >> 4) << 3);
#pragma unroll
  for (int t = 0; t < 4; ++t) {
    const int d = dBase + t * 16 + (lane & 15);
#pragma unroll
    for (int r = 0; r < 8; ++r)
      Ssum[(((size_t)(b * NCHK + j) * FDIM) + fB + r) * DIM + d] = acc[t][r];
  }
}

// ---- K3b: exclusive prefix over chunks -> Mt (transposed f16) -------------

__global__ void k_scan(const float* __restrict__ Ssum,
                       _Float16* __restrict__ Mt) { // [B][NCHK][DIM][FDIM]
  int tid = blockIdx.x * blockDim.x + threadIdx.x;
  const int total = BATCH * FDIM * DIM;
  if (tid >= total) return;
  const int b = tid / (FDIM * DIM);
  const int rem = tid - b * FDIM * DIM;
  const int f = rem / DIM, d = rem - f * DIM;
  float run = 0.0f;
#pragma unroll
  for (int j = 0; j < NCHK; ++j) {
    Mt[(((size_t)(b * NCHK + j) * DIM) + d) * FDIM + f] = (_Float16)run;
    run += Ssum[(((size_t)(b * NCHK + j) * FDIM) + f) * DIM + d];
  }
}

// ---- K3c: O = Q@M_prev + tril(Q K^T) @ V ---------------------------------
// grid (NCHK, 8, BATCH), block 256. Scores staged in LDS (f16, padded).

__global__ void k_chunk_out(const _Float16* __restrict__ Qm,
                            const _Float16* __restrict__ Km,
                            const _Float16* __restrict__ Vt,
                            const _Float16* __restrict__ Mt,
                            float* __restrict__ Ot) { // [ROWS][DIM]
  __shared__ _Float16 sc[CHUNK][CHUNK + 8];  // +8 halves pad, keeps 16B align
  const int lane = threadIdx.x & 31, wave = threadIdx.x >> 5;
  const int j = blockIdx.x, dBase = blockIdx.y * 64, b = blockIdx.z;
  const int rowTok = b * SEQ + j * CHUNK;   // first token row of chunk
  const int lBase = wave * 16;

  // Phase A: scores = Q_c K_c^T  (128x128), masked causal-inclusive
  v8f sacc[8] = {};
  for (int f0 = 0; f0 < FDIM; f0 += 32) {
    v16h a = load_frag(Qm, FDIM, rowTok + lBase, f0, lane);
#pragma unroll
    for (int tt = 0; tt < 8; ++tt) {
      v16h bf = load_frag(Km, FDIM, rowTok + tt * 16, f0, lane);
      sacc[tt] = wmma_f16(a, bf, sacc[tt]);
    }
  }
  const int mB = lBase + ((lane >> 4) << 3);
#pragma unroll
  for (int tt = 0; tt < 8; ++tt) {
    const int tl = tt * 16 + (lane & 15);
#pragma unroll
    for (int r = 0; r < 8; ++r) {
      const int ll = mB + r;
      sc[ll][tl] = (_Float16)((tl <= ll) ? sacc[tt][r] : 0.0f);
    }
  }
  __syncthreads();

  // Phase B: O = scores @ V_c + Q_c @ M_prev  (both register double-buffered)
  const _Float16* vtb = Vt + (size_t)b * DIM * SEQ;
  const _Float16* mtb = Mt + (size_t)(b * NCHK + j) * DIM * FDIM;
  v8f acc[4] = {};
  {
    v16h a_cur = load_frag(&sc[0][0], CHUNK + 8, lBase, 0, lane);
    v16h b_cur[4];
#pragma unroll
    for (int t = 0; t < 4; ++t)
      b_cur[t] = load_frag(vtb, SEQ, dBase + t * 16, j * CHUNK, lane);
    for (int i = 0; i < CHUNK / 32; ++i) {
      v16h a_next, b_next[4];
      if (i + 1 < CHUNK / 32) {
        a_next = load_frag(&sc[0][0], CHUNK + 8, lBase, (i + 1) * 32, lane);
#pragma unroll
        for (int t = 0; t < 4; ++t)
          b_next[t] = load_frag(vtb, SEQ, dBase + t * 16,
                                j * CHUNK + (i + 1) * 32, lane);
      }
#pragma unroll
      for (int t = 0; t < 4; ++t) acc[t] = wmma_f16(a_cur, b_cur[t], acc[t]);
      if (i + 1 < CHUNK / 32) {
        a_cur = a_next;
#pragma unroll
        for (int t = 0; t < 4; ++t) b_cur[t] = b_next[t];
      }
    }
  }
  {
    v16h a_cur = load_frag(Qm, FDIM, rowTok + lBase, 0, lane);
    v16h b_cur[4];
#pragma unroll
    for (int t = 0; t < 4; ++t)
      b_cur[t] = load_frag(mtb, FDIM, dBase + t * 16, 0, lane);
    for (int i = 0; i < FDIM / 32; ++i) {
      v16h a_next, b_next[4];
      if (i + 1 < FDIM / 32) {
        a_next = load_frag(Qm, FDIM, rowTok + lBase, (i + 1) * 32, lane);
#pragma unroll
        for (int t = 0; t < 4; ++t)
          b_next[t] = load_frag(mtb, FDIM, dBase + t * 16, (i + 1) * 32, lane);
      }
#pragma unroll
      for (int t = 0; t < 4; ++t) acc[t] = wmma_f16(a_cur, b_cur[t], acc[t]);
      if (i + 1 < FDIM / 32) {
        a_cur = a_next;
#pragma unroll
        for (int t = 0; t < 4; ++t) b_cur[t] = b_next[t];
      }
    }
  }
#pragma unroll
  for (int t = 0; t < 4; ++t) {
    const int d = dBase + t * 16 + (lane & 15);
#pragma unroll
    for (int r = 0; r < 8; ++r)
      Ot[(size_t)(rowTok + mB + r) * DIM + d] = acc[t][r];
  }
}

// ---- K4a: position norm + LayerNorm -> yn (f16) ---------------------------
// grid (ROWS), block 256; 2 elements per thread.

__global__ void k_layernorm(const float* __restrict__ Ot,
                            const float* __restrict__ ln_g,
                            const float* __restrict__ ln_b,
                            _Float16* __restrict__ yn) {
  __shared__ float rs[256], rq[256];
  const int row = blockIdx.x, tid = threadIdx.x;
  const int l = row & 2047;
  const float nf = rsqrtf((float)(l + 1) * (float)NPLN);
  float y0 = Ot[(size_t)row * DIM + tid] * nf;
  float y1 = Ot[(size_t)row * DIM + 256 + tid] * nf;
  rs[tid] = y0 + y1;
  rq[tid] = y0 * y0 + y1 * y1;
  __syncthreads();
#pragma unroll
  for (int s = 128; s > 0; s >>= 1) {
    if (tid < s) { rs[tid] += rs[tid + s]; rq[tid] += rq[tid + s]; }
    __syncthreads();
  }
  const float mu  = rs[0] * (1.0f / DIM);
  const float var = rq[0] * (1.0f / DIM) - mu * mu;
  const float ivs = rsqrtf(var + 1e-5f);
  yn[(size_t)row * DIM + tid]       = (_Float16)((y0 - mu) * ivs * ln_g[tid] + ln_b[tid]);
  yn[(size_t)row * DIM + 256 + tid] = (_Float16)((y1 - mu) * ivs * ln_g[256 + tid] + ln_b[256 + tid]);
}

// ---- K4b: out = x + yn @ ow + ob ------------------------------------------
// grid (32, 8), block 256. Same async LDS staging as K1.

__global__ void k_outproj(const _Float16* __restrict__ yn,
                          const _Float16* __restrict__ owt,
                          const float* __restrict__ ob,
                          const float* __restrict__ x,
                          float* __restrict__ out) {
  __shared__ _Float16 bs[2][64 * 32];
  const int tid = threadIdx.x;
  const int lane = tid & 31, wave = tid >> 5;
  const int rowBase = blockIdx.x * 128 + wave * 16;
  const int colBase = blockIdx.y * 64;

  stage_b_tile(bs[0], owt, DIM, colBase, 0, tid);
  v16h a_cur = load_frag(yn, DIM, rowBase, 0, lane);

  v8f acc[4] = {};
  for (int i = 0; i < DIM / 32; ++i) {
    stage_wait();
    __syncthreads();
    if (i + 1 < DIM / 32)
      stage_b_tile(bs[(i + 1) & 1], owt, DIM, colBase, (i + 1) * 32, tid);
    v16h a_next;
    if (i + 1 < DIM / 32)
      a_next = load_frag(yn, DIM, rowBase, (i + 1) * 32, lane);
#pragma unroll
    for (int t = 0; t < 4; ++t) {
      v16h b = load_frag(bs[i & 1], 32, t * 16, 0, lane);
      acc[t] = wmma_f16(a_cur, b, acc[t]);
    }
    a_cur = a_next;
  }

  const int mBase = rowBase + ((lane >> 4) << 3);
#pragma unroll
  for (int t = 0; t < 4; ++t) {
    const int n = colBase + t * 16 + (lane & 15);
#pragma unroll
    for (int r = 0; r < 8; ++r) {
      const size_t idx = (size_t)(mBase + r) * DIM + n;
      out[idx] = x[idx] + acc[t][r] + ob[n];
    }
  }
}

// ---------------------------------------------------------------------------

extern "C" void kernel_launch(void* const* d_in, const int* in_sizes, int n_in,
                              void* d_out, int out_size, void* d_ws, size_t ws_size,
                              hipStream_t stream) {
  const float* x   = (const float*)d_in[0];
  const float* sw  = (const float*)d_in[1];
  const float* kw1 = (const float*)d_in[2];
  const float* kb1 = (const float*)d_in[3];
  const float* kw2 = (const float*)d_in[4];
  const float* kb2 = (const float*)d_in[5];
  const float* qw1 = (const float*)d_in[6];
  const float* qb1 = (const float*)d_in[7];
  const float* qw2 = (const float*)d_in[8];
  const float* qb2 = (const float*)d_in[9];
  const float* vw  = (const float*)d_in[10];
  const float* vb  = (const float*)d_in[11];
  const float* lng = (const float*)d_in[12];
  const float* lnb = (const float*)d_in[13];
  const float* ow  = (const float*)d_in[14];
  const float* ob  = (const float*)d_in[15];
  float* out = (float*)d_out;

  // workspace carve-up (all 256B aligned)
  char* ws = (char*)d_ws;
  size_t off = 0;
  auto take = [&](size_t bytes) { char* p = ws + off; off += (bytes + 255) & ~(size_t)255; return p; };
  _Float16* xh    = (_Float16*)take((size_t)ROWS * DIM * 2);
  _Float16* hk    = (_Float16*)take((size_t)ROWS * DIM * 2);
  _Float16* hq    = (_Float16*)take((size_t)ROWS * DIM * 2);
  _Float16* vt    = (_Float16*)take((size_t)BATCH * DIM * SEQ * 2);
  _Float16* Qm    = (_Float16*)take((size_t)ROWS * FDIM * 2);
  _Float16* Km    = (_Float16*)take((size_t)ROWS * FDIM * 2);
  _Float16* Kt    = (_Float16*)take((size_t)BATCH * FDIM * SEQ * 2);
  float*    Ssum  = (float*)   take((size_t)BATCH * NCHK * FDIM * DIM * 4);
  _Float16* Mt    = (_Float16*)take((size_t)BATCH * NCHK * DIM * FDIM * 2);
  float*    Ot    = (float*)   take((size_t)ROWS * DIM * 4);
  _Float16* yn    = (_Float16*)take((size_t)ROWS * DIM * 2);
  _Float16* kw1t  = (_Float16*)take((size_t)DIM * DIM * 2);
  _Float16* qw1t  = (_Float16*)take((size_t)DIM * DIM * 2);
  _Float16* vwt   = (_Float16*)take((size_t)DIM * DIM * 2);
  _Float16* owt   = (_Float16*)take((size_t)DIM * DIM * 2);
  _Float16* kw2t  = (_Float16*)take((size_t)TP * DIM * 2);
  _Float16* qw2t  = (_Float16*)take((size_t)TP * DIM * 2);

  // P0: conversions
  k_cvt_f32_to_f16<<<(ROWS * DIM + 255) / 256, 256, 0, stream>>>(x, xh, ROWS * DIM);
  k_transpose_f16<<<(DIM * DIM + 255) / 256, 256, 0, stream>>>(kw1, kw1t, DIM, DIM);
  k_transpose_f16<<<(DIM * DIM + 255) / 256, 256, 0, stream>>>(qw1, qw1t, DIM, DIM);
  k_transpose_f16<<<(DIM * DIM + 255) / 256, 256, 0, stream>>>(vw,  vwt,  DIM, DIM);
  k_transpose_f16<<<(DIM * DIM + 255) / 256, 256, 0, stream>>>(ow,  owt,  DIM, DIM);
  k_transpose_f16<<<(DIM * TP + 255) / 256, 256, 0, stream>>>(kw2, kw2t, DIM, TP);
  k_transpose_f16<<<(DIM * TP + 255) / 256, 256, 0, stream>>>(qw2, qw2t, DIM, TP);

  // K1: Hk / Hq / Vt
  k_encode1<<<dim3(ROWS / 128, DIM / 64, 3), 256, 0, stream>>>(
      xh, kw1t, qw1t, vwt, kb1, qb1, vb, hk, hq, vt);

  // K2: phases -> Q, K, Kt
  k_phases<<<dim3(ROWS / 128), 256, 0, stream>>>(
      hk, hq, kw2t, qw2t, kb2, qb2, sw, Qm, Km, Kt);

  // K3: chunked causal linear attention
  k_chunk_kv<<<dim3(NCHK, DIM / 64, BATCH), 256, 0, stream>>>(Kt, vt, Ssum);
  k_scan<<<(BATCH * FDIM * DIM + 255) / 256, 256, 0, stream>>>(Ssum, Mt);
  k_chunk_out<<<dim3(NCHK, DIM / 64, BATCH), 256, 0, stream>>>(Qm, Km, vt, Mt, Ot);

  // K4: norm + LN + output projection + residual
  k_layernorm<<<dim3(ROWS), 256, 0, stream>>>(Ot, lng, lnb, yn);
  k_outproj<<<dim3(ROWS / 128, DIM / 64), 256, 0, stream>>>(yn, owt, ob, x, out);
}
